// MultiHeadSelfAttention_19920058319156
// MI455X (gfx1250) — matrix operand order repaired and checked by reading, not compiled, not run on hardware
//
#include <hip/hip_runtime.h>
#include <cstdint>
#include <cstddef>

typedef __attribute__((ext_vector_type(16))) __bf16        v16bf;
typedef __attribute__((ext_vector_type(8)))  float         v8f;
typedef __attribute__((ext_vector_type(4)))  unsigned int  u32x4;
typedef __attribute__((ext_vector_type(2)))  unsigned int  u32x2;
typedef __attribute__((ext_vector_type(4)))  float         f32x4;
typedef __attribute__((ext_vector_type(4)))  int           i32x4;

#ifndef __has_builtin
#define __has_builtin(x) 0
#endif

// Async global->LDS copy path (CDNA5): bypasses VGPRs, tracked by ASYNCcnt.
// Guarded so host pass / older toolchains fall back to the VGPR path.
#if __has_builtin(__builtin_amdgcn_global_load_async_to_lds_b128)
#define HAVE_ASYNC_LDS 1
#else
#define HAVE_ASYNC_LDS 0
#endif

// b128 async copy: src is v4i32* in global (addrspace 1), dst is v4i32* in LDS (addrspace 3)
#define GV4(p) ((__attribute__((address_space(1))) i32x4*)(p))
#define LV4(p) ((__attribute__((address_space(3))) i32x4*)(p))

__device__ __forceinline__ void wait_async0() {
#if __has_builtin(__builtin_amdgcn_s_wait_asynccnt)
    __builtin_amdgcn_s_wait_asynccnt(0);
#elif HAVE_ASYNC_LDS
    asm volatile("s_wait_asynccnt 0x0" ::: "memory");
#endif
}

union BFrag { v16bf v; u32x4 u[2]; };

__device__ __forceinline__ unsigned short f2bf(float f) {
    unsigned int u = __builtin_bit_cast(unsigned int, f);
    u += 0x7FFFu + ((u >> 16) & 1u);          // round-to-nearest-even
    return (unsigned short)(u >> 16);
}

__device__ __forceinline__ v8f wmma_bf16(const BFrag& a, const BFrag& b, v8f c) {
    // D = A(16x32 bf16) x B(32x16 bf16) + C(16x16 f32)
    return __builtin_amdgcn_wmma_f32_16x16x32_bf16(
        /*neg_a=*/false, a.v, /*neg_b=*/false, b.v,
        /*c_mod=*/(short)0, c, /*reuse_a=*/false, /*reuse_b=*/false);
}

constexpr int Hdim  = 1024;
constexpr int Sdim  = 2048;
constexpr int Bdim  = 2;
constexpr int NHEAD = 16;
constexpr int DK    = 64;
constexpr int Mtot  = Bdim * Sdim;          // 4096

// ---------------- GEMM: Y = X @ W^T + b ------------------------------------
// Block tile 128x128, K-step 32, 8 waves (2x4), each wave 64x32 = 4x2 WMMA accs.
constexpr int BM  = 128;
constexpr int BN  = 128;
constexpr int BK  = 32;
constexpr int LDT = BK + 8;                 // 40 halfwords: conflict-free b128 reads

template<bool A_BF16, bool OUT_HEADS>
__global__ __launch_bounds__(256) void gemm_wb(
    const void* __restrict__ Ap, const float* __restrict__ W,
    const float* __restrict__ bias, void* __restrict__ Out)
{
    __shared__ __attribute__((aligned(16))) unsigned short As[BM * LDT];
    __shared__ __attribute__((aligned(16))) unsigned short Bs[BN * LDT];

    const int tid = threadIdx.x, lane = tid & 31, wv = tid >> 5;
    const int lrow = lane & 15, lhalf = lane >> 4;
    const int wm = (wv >> 2) * 64, wn = (wv & 3) * 32;
    const int m0 = blockIdx.x * BM, n0 = blockIdx.y * BN;

    const v8f vzero = {0.f,0.f,0.f,0.f,0.f,0.f,0.f,0.f};
    v8f acc[4][2];
    #pragma unroll
    for (int i = 0; i < 4; ++i)
        #pragma unroll
        for (int j = 0; j < 2; ++j) acc[i][j] = vzero;

    for (int k0 = 0; k0 < Hdim; k0 += BK) {
        __syncthreads();
        // ---- A tile ----
        if constexpr (A_BF16) {
#if HAVE_ASYNC_LDS
            // bf16 A goes to LDS unchanged: async DMA, no VGPR round trip.
            #pragma unroll
            for (int i = 0; i < 2; ++i) {             // 512 16B chunks / 256 thr
                int c = tid + i * 256, r = c >> 2, kc = (c & 3) * 8;
                __builtin_amdgcn_global_load_async_to_lds_b128(
                    GV4((const unsigned short*)Ap + (size_t)(m0 + r) * Hdim + k0 + kc),
                    LV4(&As[r * LDT + kc]), 0, 0);
            }
#else
            #pragma unroll
            for (int i = 0; i < 4; ++i) {
                int c = tid + i * 256, r = c >> 3, kc = (c & 7) * 4;
                const unsigned short* A = (const unsigned short*)Ap;
                u32x2 d = *(const u32x2*)(A + (size_t)(m0 + r) * Hdim + k0 + kc);
                *(u32x2*)&As[r * LDT + kc] = d;
            }
#endif
        } else {
            #pragma unroll
            for (int i = 0; i < 4; ++i) {             // fp32 -> bf16 convert on the fly
                int c = tid + i * 256, r = c >> 3, kc = (c & 7) * 4;
                const float* A = (const float*)Ap;
                f32x4 d = *(const f32x4*)(A + (size_t)(m0 + r) * Hdim + k0 + kc);
                u32x2 p;
                p.x = (unsigned)f2bf(d.x) | ((unsigned)f2bf(d.y) << 16);
                p.y = (unsigned)f2bf(d.z) | ((unsigned)f2bf(d.w) << 16);
                *(u32x2*)&As[r * LDT + kc] = p;
            }
        }
        // ---- W tile (always fp32 -> bf16) ----
        #pragma unroll
        for (int i = 0; i < 4; ++i) {
            int c = tid + i * 256, r = c >> 3, kc = (c & 7) * 4;
            f32x4 wd = *(const f32x4*)(W + (size_t)(n0 + r) * Hdim + k0 + kc);
            u32x2 wp;
            wp.x = (unsigned)f2bf(wd.x) | ((unsigned)f2bf(wd.y) << 16);
            wp.y = (unsigned)f2bf(wd.z) | ((unsigned)f2bf(wd.w) << 16);
            *(u32x2*)&Bs[r * LDT + kc] = wp;
        }
#if HAVE_ASYNC_LDS
        if constexpr (A_BF16) wait_async0();
#endif
        __syncthreads();

        BFrag bfr[2];
        #pragma unroll
        for (int nt = 0; nt < 2; ++nt) {              // B: lane=col, k split 0..7/16..23
            const unsigned short* bp = &Bs[(wn + nt * 16 + lrow) * LDT + lhalf * 8];
            bfr[nt].u[0] = *(const u32x4*)bp;
            bfr[nt].u[1] = *(const u32x4*)(bp + 16);
        }
        #pragma unroll
        for (int mt = 0; mt < 4; ++mt) {              // A: lane=row, k split 0..7/16..23
            BFrag af;
            const unsigned short* ap = &As[(wm + mt * 16 + lrow) * LDT + lhalf * 8];
            af.u[0] = *(const u32x4*)ap;
            af.u[1] = *(const u32x4*)(ap + 16);
            #pragma unroll
            for (int nt = 0; nt < 2; ++nt)
                acc[mt][nt] = wmma_bf16(af, bfr[nt], acc[mt][nt]);
        }
    }

    #pragma unroll
    for (int mt = 0; mt < 4; ++mt) {
        #pragma unroll
        for (int nt = 0; nt < 2; ++nt) {
            int col = n0 + wn + nt * 16 + lrow;
            float bv = bias[col];
            #pragma unroll
            for (int r = 0; r < 8; ++r) {
                int row = m0 + wm + mt * 16 + r + (lhalf ? 8 : 0);
                float val = acc[mt][nt][r] + bv;
                if (OUT_HEADS) {      // bf16, head-split (b,h,s,d)
                    int h = col >> 6, d = col & 63, b = row >> 11, s = row & 2047;
                    ((unsigned short*)Out)[((size_t)((b * NHEAD + h) * Sdim + s)) * DK + d] = f2bf(val);
                } else {              // fp32 flat (b*s, hidden)
                    ((float*)Out)[(size_t)row * Hdim + col] = val;
                }
            }
        }
    }
}

// ---------------- Flash attention ------------------------------------------
// Block: 128 thr (4 waves), 64 query rows (16 per wave), key tiles of 64.
constexpr int LDK2 = 72;                    // padded LDS stride, conflict-free

__global__ __launch_bounds__(128) void attn_fwd(
    const unsigned short* __restrict__ qh, const unsigned short* __restrict__ kh,
    const unsigned short* __restrict__ vh, const int* __restrict__ mask,
    unsigned short* __restrict__ o)
{
    __shared__ __attribute__((aligned(16))) unsigned short Kt[64 * LDK2]; // [key][d]
    __shared__ __attribute__((aligned(16))) unsigned short Vt[64 * LDK2]; // [d][key]
    __shared__ __attribute__((aligned(16))) unsigned short Pb[64 * LDK2]; // [q][key]

    const int tid = threadIdx.x, lane = tid & 31, wv = tid >> 5;
    const int lrow = lane & 15, lhalf = lane >> 4;
    const int bh = blockIdx.y, b = bh >> 4, h = bh & 15;
    const int q0 = blockIdx.x * 64;

    // Q fragments straight from global (A-layout: lane=row, two b128 per frag)
    const unsigned short* qb = qh + ((size_t)bh * Sdim + q0 + wv * 16 + lrow) * DK;
    BFrag qf[2];
    qf[0].u[0] = *(const u32x4*)(qb + lhalf * 8);
    qf[0].u[1] = *(const u32x4*)(qb + 16 + lhalf * 8);
    qf[1].u[0] = *(const u32x4*)(qb + 32 + lhalf * 8);
    qf[1].u[1] = *(const u32x4*)(qb + 48 + lhalf * 8);

    const v8f vzero = {0.f,0.f,0.f,0.f,0.f,0.f,0.f,0.f};
    v8f oacc[4];
    #pragma unroll
    for (int dt = 0; dt < 4; ++dt) oacc[dt] = vzero;
    float mrow[8], lsum[8];
    #pragma unroll
    for (int r = 0; r < 8; ++r) { mrow[r] = -1e30f; lsum[r] = 0.f; }

    const unsigned short* kb = kh + (size_t)bh * Sdim * DK;
    const unsigned short* vb = vh + (size_t)bh * Sdim * DK;
    const int* mb = mask + (size_t)b * Sdim * Sdim;

    for (int t0 = 0; t0 < Sdim; t0 += 64) {
        __syncthreads();
        #pragma unroll
        for (int i = 0; i < 4; ++i) {                 // 512 8-elem chunks / 128 thr
            int c = tid + i * 128, r = c >> 3, dc = (c & 7) * 8;
#if HAVE_ASYNC_LDS
            // K tile: layout-preserving copy -> async DMA into LDS.
            __builtin_amdgcn_global_load_async_to_lds_b128(
                GV4(kb + (size_t)(t0 + r) * DK + dc),
                LV4(&Kt[r * LDK2 + dc]), 0, 0);
#else
            u32x4 kd = *(const u32x4*)(kb + (size_t)(t0 + r) * DK + dc);
            *(u32x4*)&Kt[r * LDK2 + dc] = kd;
#endif
            // V tile: transposed on the way in (must go through VGPRs).
            u32x4 vd = *(const u32x4*)(vb + (size_t)(t0 + r) * DK + dc);
            const unsigned short* vs = (const unsigned short*)&vd;
            #pragma unroll
            for (int j = 0; j < 8; ++j) Vt[(dc + j) * LDK2 + r] = vs[j];
        }
#if HAVE_ASYNC_LDS
        wait_async0();
#endif
        __syncthreads();

        // S = Q K^T (16 q-rows x 64 keys per wave)
        v8f sacc[4];
        #pragma unroll
        for (int nt = 0; nt < 4; ++nt) sacc[nt] = vzero;
        #pragma unroll
        for (int ds_ = 0; ds_ < 2; ++ds_) {
            #pragma unroll
            for (int nt = 0; nt < 4; ++nt) {
                BFrag kf;   // B-layout: lane=key col, d split 0..7/16..23
                const unsigned short* kp = &Kt[(nt * 16 + lrow) * LDK2 + ds_ * 32 + lhalf * 8];
                kf.u[0] = *(const u32x4*)kp;
                kf.u[1] = *(const u32x4*)(kp + 16);
                sacc[nt] = wmma_bf16(qf[ds_], kf, sacc[nt]);
            }
        }

        // scale, mask (masked logit -> 1e-9 per reference), online softmax
        float x[4][8];
        #pragma unroll
        for (int nt = 0; nt < 4; ++nt) {
            int key = t0 + nt * 16 + lrow;
            #pragma unroll
            for (int r = 0; r < 8; ++r) {
                int qr = q0 + wv * 16 + r + (lhalf ? 8 : 0);
                int mvv = mb[(size_t)qr * Sdim + key];
                float s = sacc[nt][r] * 0.125f;       // 1/sqrt(64)
                x[nt][r] = (mvv == 0) ? 1e-9f : s;
            }
        }
        #pragma unroll
        for (int r = 0; r < 8; ++r) {
            float mx = fmaxf(fmaxf(x[0][r], x[1][r]), fmaxf(x[2][r], x[3][r]));
            #pragma unroll
            for (int off = 1; off < 16; off <<= 1)    // reduce across 16-lane half
                mx = fmaxf(mx, __shfl_xor(mx, off, 32));
            float mn = fmaxf(mrow[r], mx);
            float corr = __expf(mrow[r] - mn);
            float sum = 0.f;
            #pragma unroll
            for (int nt = 0; nt < 4; ++nt) {
                float p = __expf(x[nt][r] - mn);
                x[nt][r] = p; sum += p;
            }
            #pragma unroll
            for (int off = 1; off < 16; off <<= 1)
                sum += __shfl_xor(sum, off, 32);
            lsum[r] = lsum[r] * corr + sum;
            mrow[r] = mn;
            #pragma unroll
            for (int dt = 0; dt < 4; ++dt) oacc[dt][r] *= corr;
        }

        // P -> LDS as bf16 (re-layout C-layout -> A-layout)
        #pragma unroll
        for (int nt = 0; nt < 4; ++nt)
            #pragma unroll
            for (int r = 0; r < 8; ++r)
                Pb[(wv * 16 + r + (lhalf ? 8 : 0)) * LDK2 + nt * 16 + lrow] = f2bf(x[nt][r]);
        __syncthreads();

        // O += P V
        #pragma unroll
        for (int ks = 0; ks < 2; ++ks) {
            BFrag pf;
            const unsigned short* pp = &Pb[(wv * 16 + lrow) * LDK2 + ks * 32 + lhalf * 8];
            pf.u[0] = *(const u32x4*)pp;
            pf.u[1] = *(const u32x4*)(pp + 16);
            #pragma unroll
            for (int dt = 0; dt < 4; ++dt) {
                BFrag vf;   // B-layout from transposed V: lane=d col, keys split 0..7/16..23
                const unsigned short* vp = &Vt[(dt * 16 + lrow) * LDK2 + ks * 32 + lhalf * 8];
                vf.u[0] = *(const u32x4*)vp;
                vf.u[1] = *(const u32x4*)(vp + 16);
                oacc[dt] = wmma_bf16(pf, vf, oacc[dt]);
            }
        }
    }

    // normalize and store O as bf16 into (b,s,hidden) layout
    #pragma unroll
    for (int dt = 0; dt < 4; ++dt) {
        int col = h * DK + dt * 16 + lrow;
        #pragma unroll
        for (int r = 0; r < 8; ++r) {
            int qr = q0 + wv * 16 + r + (lhalf ? 8 : 0);
            o[((size_t)b * Sdim + qr) * Hdim + col] = f2bf(oacc[dt][r] / lsum[r]);
        }
    }
}

// ---------------- Launch ----------------------------------------------------
extern "C" void kernel_launch(void* const* d_in, const int* in_sizes, int n_in,
                              void* d_out, int out_size, void* d_ws, size_t ws_size,
                              hipStream_t stream)
{
    const float* q    = (const float*)d_in[0];
    const float* k    = (const float*)d_in[1];
    const float* v    = (const float*)d_in[2];
    const int*   mask = (const int*)  d_in[3];
    const float* Wq   = (const float*)d_in[4];
    const float* bq   = (const float*)d_in[5];
    const float* Wk   = (const float*)d_in[6];
    const float* bk   = (const float*)d_in[7];
    const float* Wv   = (const float*)d_in[8];
    const float* bv   = (const float*)d_in[9];
    const float* Wo   = (const float*)d_in[10];
    const float* bo   = (const float*)d_in[11];

    // workspace: 4 bf16 buffers of B*H*S*DK elements (8 MB each, 32 MB total)
    const size_t per = (size_t)Bdim * NHEAD * Sdim * DK;     // 4,194,304
    unsigned short* qh = (unsigned short*)d_ws;
    unsigned short* kh = qh + per;
    unsigned short* vh = kh + per;
    unsigned short* oo = vh + per;

    dim3 ggrid(Mtot / BM, Hdim / BN);                        // (32, 8)
    gemm_wb<false, true><<<ggrid, 256, 0, stream>>>(q, Wq, bq, qh);
    gemm_wb<false, true><<<ggrid, 256, 0, stream>>>(k, Wk, bk, kh);
    gemm_wb<false, true><<<ggrid, 256, 0, stream>>>(v, Wv, bv, vh);

    dim3 agrid(Sdim / 64, Bdim * NHEAD);                     // (32, 32)
    attn_fwd<<<agrid, 128, 0, stream>>>(qh, kh, vh, mask, oo);

    gemm_wb<true, false><<<ggrid, 256, 0, stream>>>(oo, Wo, bo, (float*)d_out);

    (void)in_sizes; (void)n_in; (void)out_size; (void)ws_size;
}
